// Head_42511586296094
// MI455X (gfx1250) — matrix operand order repaired and checked by reading, not compiled, not run on hardware
//
#include <hip/hip_runtime.h>
#include <hip/hip_bf16.h>

// Fused attention head, all matmuls on bf16 WMMA with async (ASYNCcnt)
// double-buffered LDS staging.
// Workspace (requires ~70 MiB):
//   Q   bf16 [B=4][H=16][S=2048][D=64]  (pre-scaled by 1/sqrt(64))
//   K   bf16 [B][H][S][D]
//   Vt  bf16 [B][H][D][S]   (transposed: contiguous B-fragments for P.V)
//   xbf bf16 [8192][1024]   (x converted once)
//   Wt  bf16 [3][1024][1024] (Wq/Wk/Wv converted + transposed once: [out][in])

typedef __bf16 bf16;
typedef __attribute__((ext_vector_type(16))) __bf16 v16bf;
typedef __attribute__((ext_vector_type(8)))  __bf16 v8bf;
typedef __attribute__((ext_vector_type(4)))  __bf16 v4bf;
typedef __attribute__((ext_vector_type(2)))  __bf16 v2bf;
typedef __attribute__((ext_vector_type(8)))  float   v8f;
typedef __attribute__((ext_vector_type(4)))  float   v4f;

#define EMB   1024
#define HEADS 16
#define DH    64
#define SEQ   2048
#define BATCH 4

__device__ __forceinline__ bf16 f2bf(float f) {
  unsigned u = __builtin_bit_cast(unsigned, f);
  u += 0x7FFFu + ((u >> 16) & 1u);          // round-to-nearest-even
  unsigned short h = (unsigned short)(u >> 16);
  return __builtin_bit_cast(bf16, h);
}

// Packed f32x2 -> bf16x2 (native v_cvt_pk_bf16_f32 when available).
#if __has_builtin(__builtin_amdgcn_cvt_pk_bf16_f32)
__device__ __forceinline__ v2bf pk2(float a, float b) {
  return __builtin_amdgcn_cvt_pk_bf16_f32(a, b);
}
#else
__device__ __forceinline__ v2bf pk2(float a, float b) {
  v2bf r; r[0] = f2bf(a); r[1] = f2bf(b); return r;
}
#endif

__device__ __forceinline__ v16bf cat8(v8bf lo, v8bf hi) {
  return __builtin_shufflevector(lo, hi, 0,1,2,3,4,5,6,7,8,9,10,11,12,13,14,15);
}

__device__ __forceinline__ v8f wmma_bf16(v16bf a, v16bf b, v8f c) {
  return __builtin_amdgcn_wmma_f32_16x16x32_bf16(false, a, false, b,
                                                 (short)0, c, false, false);
}

// LDS byte offset of a generic shared pointer (flat->local = low 32 bits).
__device__ __forceinline__ unsigned lds_off(const void* p) {
  return (unsigned)(unsigned long long)p;
}

// GLOBAL_LOAD_ASYNC_TO_LDS_B128: per-lane 16B memory->LDS, tracked by ASYNCcnt.
__device__ __forceinline__ void async_b128(unsigned lds_addr, const void* gptr) {
  asm volatile("global_load_async_to_lds_b128 %0, %1, off"
               :: "v"(lds_addr), "v"(gptr)
               : "memory");
}
__device__ __forceinline__ void wait_async0() {
  asm volatile("s_wait_asynccnt 0x0" ::: "memory");
}

// ---------------------------------------------------------------------------
// Pre-pass 0a: x f32 -> bf16, streaming.
// ---------------------------------------------------------------------------
__global__ __launch_bounds__(256)
void convert_x_kernel(const float* __restrict__ x, bf16* __restrict__ xbf) {
  int idx = (blockIdx.x * 256 + threadIdx.x) * 4;
  v4f d = *(const v4f*)(x + idx);
  v2bf a = pk2(d[0], d[1]);
  v2bf b = pk2(d[2], d[3]);
  *(v4bf*)(xbf + idx) = __builtin_shufflevector(a, b, 0, 1, 2, 3);
}

// ---------------------------------------------------------------------------
// Pre-pass 0b: W f32 [in][out] -> Wt bf16 [out][in], 64x64 LDS-tiled transpose.
// ---------------------------------------------------------------------------
__global__ __launch_bounds__(256)
void convert_wt_kernel(const float* __restrict__ Wq, const float* __restrict__ Wk,
                       const float* __restrict__ Wv, bf16* __restrict__ Wt) {
  __shared__ bf16 t[64][68];
  const int z = blockIdx.z;
  const float* W = (z == 0) ? Wq : (z == 1) ? Wk : Wv;
  bf16* dst = Wt + (size_t)z * EMB * EMB;
  const int k0 = blockIdx.x * 64, n0 = blockIdx.y * 64;
  const int tid = threadIdx.x;
#pragma unroll
  for (int i = 0; i < 4; ++i) {
    int idx = tid + i * 256;                   // 0..1023 over 64x16 float4
    int r = idx >> 4, c4 = (idx & 15) * 4;
    v4f d = *(const v4f*)(W + (size_t)(k0 + r) * EMB + n0 + c4);
    *(v2bf*)&t[r][c4]     = pk2(d[0], d[1]);
    *(v2bf*)&t[r][c4 + 2] = pk2(d[2], d[3]);
  }
  __syncthreads();
#pragma unroll
  for (int i = 0; i < 4; ++i) {
    int idx = tid + i * 256;
    int rr = idx >> 4, c4 = (idx & 15) * 4;    // rr = local n, c4 = local k
    v4bf o;
    o[0] = t[c4 + 0][rr]; o[1] = t[c4 + 1][rr];
    o[2] = t[c4 + 2][rr]; o[3] = t[c4 + 3][rr];
    *(v4bf*)(dst + (size_t)(n0 + rr) * EMB + k0 + c4) = o;
  }
}

// ---------------------------------------------------------------------------
// Kernel 1: C[8192,1024] = xbf @ Wt^T + b, bf16 out. blockIdx.z selects Q/K/V.
// WG = 256 threads (8 waves), tile 128(M) x 128(N), K-step 32, double-buffered
// async LDS staging (pure byte copies, no VALU in the hot loop).
// ---------------------------------------------------------------------------
__global__ __launch_bounds__(256)
void qkv_proj_kernel(const bf16* __restrict__ xbf, const bf16* __restrict__ Wt,
                     const float* __restrict__ bq, const float* __restrict__ bk,
                     const float* __restrict__ bv,
                     bf16* __restrict__ Q, bf16* __restrict__ K,
                     bf16* __restrict__ Vt) {
  __shared__ __align__(16) bf16 lA[2][128][40];   // [buf][m][k]
  __shared__ __align__(16) bf16 lB[2][128][40];   // [buf][n][k]

  const int tid  = threadIdx.x;
  const int lane = tid & 31;
  const int w    = tid >> 5;
  const int wm   = w >> 1;                     // 0..3
  const int wn   = w & 1;                      // 0..1
  const int l15  = lane & 15;
  const int hs   = lane >> 4;
  const int kbA  = hs * 8;                     // A-frag: K = kbA..+7, kbA+16..+23
  const int kbB  = hs * 8;                     // B-frag: K = kbB..kbB+7, kbB+16..+23
  const int m0   = blockIdx.x * 128;
  const int n0   = blockIdx.y * 128;
  const int z    = blockIdx.z;

  const bf16*  Wm   = Wt + (size_t)z * EMB * EMB;
  const float* bias = (z == 0) ? bq : (z == 1) ? bk : bv;

  v8f acc[2][4];
#pragma unroll
  for (int mi = 0; mi < 2; ++mi)
#pragma unroll
    for (int j = 0; j < 4; ++j) acc[mi][j] = (v8f){};

  // Async staging of one 128x32 A tile + 128x32 B tile into buffer `buf`.
  auto stage = [&](int k0, int buf) {
#pragma unroll
    for (int i = 0; i < 2; ++i) {
      int c   = tid + i * 256;                 // 0..511 over 128 rows x 4 chunks
      int row = c >> 2;
      int q8  = (c & 3) * 8;                   // 8 bf16 = 16B chunk
      async_b128(lds_off(&lA[buf][row][q8]),
                 xbf + (size_t)(m0 + row) * EMB + k0 + q8);
      async_b128(lds_off(&lB[buf][row][q8]),
                 Wm + (size_t)(n0 + row) * EMB + k0 + q8);
    }
  };

  stage(0, 0);                                 // prologue

  for (int k0 = 0; k0 < EMB; k0 += 32) {
    const int cur = (k0 >> 5) & 1;
    wait_async0();                             // my chunks landed
    __syncthreads();                           // all landed; prev buffer free
    if (k0 + 32 < EMB) stage(k0 + 32, cur ^ 1);

    v16bf a[2], b[4];
#pragma unroll
    for (int mi = 0; mi < 2; ++mi) {
      int row = wm * 32 + mi * 16 + l15;
      a[mi] = cat8(*(const v8bf*)&lA[cur][row][kbA],
                   *(const v8bf*)&lA[cur][row][kbA + 16]);
    }
#pragma unroll
    for (int j = 0; j < 4; ++j) {
      int col = wn * 64 + j * 16 + l15;
      b[j] = cat8(*(const v8bf*)&lB[cur][col][kbB],
                  *(const v8bf*)&lB[cur][col][kbB + 16]);
    }
#pragma unroll
    for (int mi = 0; mi < 2; ++mi)
#pragma unroll
      for (int j = 0; j < 4; ++j)
        acc[mi][j] = wmma_bf16(a[mi], b[j], acc[mi][j]);
    // No trailing barrier: top-of-loop wait+barrier gates buffer reuse.
  }

  // Epilogue: + bias, (Q only) * 1/sqrt(64), convert bf16, scatter to layout.
  const int   rofs  = hs * 8;
  const float scale = (z == 0) ? 0.125f : 1.0f;
#pragma unroll
  for (int mi = 0; mi < 2; ++mi) {
#pragma unroll
    for (int j = 0; j < 4; ++j) {
      int n = n0 + wn * 64 + j * 16 + l15;
      float bb = bias[n];
      int h = n >> 6, d = n & 63;
      if (z == 2) {
        // Vt[b][h][d][s]: consecutive r -> consecutive s, pack pairs.
#pragma unroll
        for (int r = 0; r < 8; r += 2) {
          int m = m0 + wm * 32 + mi * 16 + r + rofs;
          int bidx = m >> 11, s = m & (SEQ - 1);
          v2bf pv = pk2(acc[mi][j][r] + bb, acc[mi][j][r + 1] + bb);
          *(v2bf*)&Vt[(((size_t)bidx * HEADS + h) * DH + d) * SEQ + s] = pv;
        }
      } else {
        bf16* dst = (z == 0) ? Q : K;
#pragma unroll
        for (int r = 0; r < 8; r += 2) {
          int m = m0 + wm * 32 + mi * 16 + r + rofs;
          int bidx = m >> 11, s = m & (SEQ - 1);
          v2bf pv = pk2((acc[mi][j][r] + bb) * scale,
                        (acc[mi][j][r + 1] + bb) * scale);
          size_t base = (((size_t)bidx * HEADS + h) * SEQ + s) * DH + d;
          dst[base]      = pv[0];
          dst[base + DH] = pv[1];               // s+1 -> +DH in [s][d] layout
        }
      }
    }
  }
}

// ---------------------------------------------------------------------------
// Kernel 2: flash attention. WG per (bh, 128-row s-block); 8 waves each own
// 16 Q rows. t-tiles of 64 with double-buffered async LDS staging.
// ---------------------------------------------------------------------------
__global__ __launch_bounds__(256)
void attn_kernel(const bf16* __restrict__ Q, const bf16* __restrict__ K,
                 const bf16* __restrict__ Vt, float* __restrict__ out) {
  __shared__ __align__(16) bf16 lK[2][64][72];    // [buf][t][d]
  __shared__ __align__(16) bf16 lV[2][64][72];    // [buf][d][t]
  __shared__ __align__(16) bf16 lP[8][16][72];    // per-wave P scratch

  const int tid  = threadIdx.x;
  const int lane = tid & 31;
  const int w    = tid >> 5;
  const int l15  = lane & 15;
  const int hs   = lane >> 4;
  const int kbA  = hs * 8;                        // A-fragment K-base
  const int kbB  = hs * 8;                        // B-fragment K-base (same as A)

  const int bh = blockIdx.y;                      // 0..63
  const int b  = bh >> 4, h = bh & 15;
  const int sblock = blockIdx.x * 128;
  const int srow   = sblock + w * 16 + l15;

  const bf16* Qb = Q  + (size_t)bh * SEQ * DH;
  const bf16* Kb = K  + (size_t)bh * SEQ * DH;
  const bf16* Vb = Vt + (size_t)bh * DH * SEQ;

  // Q fragments (16 rows x 64 d -> two 16x32 A-fragments), scale pre-folded.
  v16bf q0 = cat8(*(const v8bf*)(Qb + (size_t)srow * DH + kbA),
                  *(const v8bf*)(Qb + (size_t)srow * DH + kbA + 16));
  v16bf q1 = cat8(*(const v8bf*)(Qb + (size_t)srow * DH + 32 + kbA),
                  *(const v8bf*)(Qb + (size_t)srow * DH + 32 + kbA + 16));

  v8f o[4];
#pragma unroll
  for (int j = 0; j < 4; ++j) o[j] = (v8f){};
  float mrow[8], lrow[8];
#pragma unroll
  for (int r = 0; r < 8; ++r) { mrow[r] = -1e30f; lrow[r] = 0.f; }

  const float LOG2E = 1.4426950408889634f;

  // Async staging of one 64x64 K tile + 64x64 Vt tile into buffer `buf`.
  auto stage = [&](int tb, int buf) {
    int tr  = tid >> 2;                           // 0..63
    int c16 = (tid & 3) * 16;
    async_b128(lds_off(&lK[buf][tr][c16]),
               Kb + (size_t)(tb + tr) * DH + c16);
    async_b128(lds_off(&lK[buf][tr][c16 + 8]),
               Kb + (size_t)(tb + tr) * DH + c16 + 8);
    async_b128(lds_off(&lV[buf][tr][c16]),
               Vb + (size_t)tr * SEQ + tb + c16);
    async_b128(lds_off(&lV[buf][tr][c16 + 8]),
               Vb + (size_t)tr * SEQ + tb + c16 + 8);
  };

  stage(0, 0);                                    // prologue

  for (int tb = 0; tb < SEQ; tb += 64) {
    const int cur = (tb >> 6) & 1;
    wait_async0();                                // my tile-chunk landed
    __syncthreads();                              // everyone's landed; prev buf free
    if (tb + 64 < SEQ) stage(tb + 64, cur ^ 1);   // overlap next tile with compute

    // Scores: S[16 x 64] = Q . K^T  (B-frag in same K order as A-frag)
    v8f st[4];
#pragma unroll
    for (int j = 0; j < 4; ++j) {
      int tcol = j * 16 + l15;
      v16bf bk0 = cat8(*(const v8bf*)&lK[cur][tcol][kbB],
                       *(const v8bf*)&lK[cur][tcol][kbB + 16]);
      v16bf bk1 = cat8(*(const v8bf*)&lK[cur][tcol][32 + kbB],
                       *(const v8bf*)&lK[cur][tcol][32 + kbB + 16]);
      v8f zz = (v8f){};
      zz = wmma_bf16(q0, bk0, zz);
      zz = wmma_bf16(q1, bk1, zz);
      st[j] = zz;
    }

    // Online softmax. Row r lives across 16 lanes of a half-wave.
#pragma unroll
    for (int r = 0; r < 8; ++r) {
      float mx = fmaxf(fmaxf(st[0][r], st[1][r]), fmaxf(st[2][r], st[3][r]));
      mx = fmaxf(mx, __shfl_xor(mx, 1));
      mx = fmaxf(mx, __shfl_xor(mx, 2));
      mx = fmaxf(mx, __shfl_xor(mx, 4));
      mx = fmaxf(mx, __shfl_xor(mx, 8));
      float mnew  = fmaxf(mrow[r], mx);
      float alpha = exp2f((mrow[r] - mnew) * LOG2E);
      mrow[r] = mnew;
      float p0v = exp2f((st[0][r] - mnew) * LOG2E);
      float p1v = exp2f((st[1][r] - mnew) * LOG2E);
      float p2v = exp2f((st[2][r] - mnew) * LOG2E);
      float p3v = exp2f((st[3][r] - mnew) * LOG2E);
      float rsum = (p0v + p1v) + (p2v + p3v);
      int prow = r + hs * 8;
      v2bf pa = pk2(p0v, p1v);
      v2bf pb = pk2(p2v, p3v);
      lP[w][prow][0 * 16 + l15] = pa[0];
      lP[w][prow][1 * 16 + l15] = pa[1];
      lP[w][prow][2 * 16 + l15] = pb[0];
      lP[w][prow][3 * 16 + l15] = pb[1];
      rsum += __shfl_xor(rsum, 1);
      rsum += __shfl_xor(rsum, 2);
      rsum += __shfl_xor(rsum, 4);
      rsum += __shfl_xor(rsum, 8);
      lrow[r] = lrow[r] * alpha + rsum;
#pragma unroll
      for (int j = 0; j < 4; ++j) o[j][r] *= alpha;
    }

    // P fragments from per-wave LDS (same-wave LDS ops are in-order: no barrier)
    v16bf p0 = cat8(*(const v8bf*)&lP[w][l15][kbA],
                    *(const v8bf*)&lP[w][l15][kbA + 16]);
    v16bf p1 = cat8(*(const v8bf*)&lP[w][l15][32 + kbA],
                    *(const v8bf*)&lP[w][l15][32 + kbA + 16]);

    // O += P . V   (B-frag from lV[dcol][t], same K order as A-frag)
#pragma unroll
    for (int j = 0; j < 4; ++j) {
      int dcol = j * 16 + l15;
      v16bf bv0 = cat8(*(const v8bf*)&lV[cur][dcol][kbB],
                       *(const v8bf*)&lV[cur][dcol][kbB + 16]);
      v16bf bv1 = cat8(*(const v8bf*)&lV[cur][dcol][32 + kbB],
                       *(const v8bf*)&lV[cur][dcol][32 + kbB + 16]);
      o[j] = wmma_bf16(p0, bv0, o[j]);
      o[j] = wmma_bf16(p1, bv1, o[j]);
    }
    // No trailing barrier: top-of-loop wait+barrier gates buffer reuse.
  }

  // Epilogue: normalize and write fp32 out[b][s][h*64+d]
#pragma unroll
  for (int r = 0; r < 8; ++r) {
    float inv = 1.0f / lrow[r];
    int s = sblock + w * 16 + r + hs * 8;
#pragma unroll
    for (int j = 0; j < 4; ++j) {
      int e = h * DH + j * 16 + l15;
      out[((size_t)b * SEQ + s) * EMB + e] = o[j][r] * inv;
    }
  }
}

extern "C" void kernel_launch(void* const* d_in, const int* in_sizes, int n_in,
                              void* d_out, int out_size, void* d_ws, size_t ws_size,
                              hipStream_t stream) {
  const float* x  = (const float*)d_in[0];
  const float* Wq = (const float*)d_in[1];
  const float* bq = (const float*)d_in[2];
  const float* Wk = (const float*)d_in[3];
  const float* bk = (const float*)d_in[4];
  const float* Wv = (const float*)d_in[5];
  const float* bv = (const float*)d_in[6];
  float* out = (float*)d_out;

  const size_t qkvElems = (size_t)BATCH * HEADS * SEQ * DH;  // 8,388,608
  bf16* Qws = (bf16*)d_ws;
  bf16* Kws = Qws + qkvElems;
  bf16* Vws = Kws + qkvElems;                    // transposed [b][h][d][s]
  bf16* xbf = Vws + qkvElems;                    // [8192][1024]
  bf16* Wt  = xbf + (size_t)BATCH * SEQ * EMB;   // [3][1024][1024]

  convert_x_kernel<<<(BATCH * SEQ * EMB) / (256 * 4), 256, 0, stream>>>(x, xbf);
  convert_wt_kernel<<<dim3(EMB / 64, EMB / 64, 3), 256, 0, stream>>>(Wq, Wk, Wv, Wt);

  dim3 g1((BATCH * SEQ) / 128, EMB / 128, 3);    // 64 x 8 x 3
  qkv_proj_kernel<<<g1, 256, 0, stream>>>(xbf, Wt, bq, bk, bv, Qws, Kws, Vws);

  dim3 g2(SEQ / 128, BATCH * HEADS);             // 16 x 64
  attn_kernel<<<g2, 256, 0, stream>>>(Qws, Kws, Vws, out);
}
